// CrossModalAttention_1760936592029
// MI455X (gfx1250) — matrix operand order repaired and checked by reading, not compiled, not run on hardware
//
#include <hip/hip_runtime.h>
#include <stdint.h>

// ---------------------------------------------------------------- types
typedef __bf16 bf16;
typedef __attribute__((ext_vector_type(16))) __bf16 v16bf;
typedef __attribute__((ext_vector_type(8)))  __bf16 v8bf;
typedef __attribute__((ext_vector_type(8)))  float  v8f;
typedef __attribute__((ext_vector_type(4)))  unsigned v4u;
typedef __attribute__((ext_vector_type(8)))  unsigned v8u;

#define HW 2304
#define CD 512
#define NB 16
#define NHEAD 8
#define ATTN_SCALE 0.125f   // 64^-0.5

static __device__ __forceinline__ bf16 f2bf(float x) { return (bf16)x; }

template <typename T> struct alignas(4 * sizeof(T)) V4 { T v[4]; };

// ----------------------------------------------- CDNA5 async data movers
// per-lane 16B global -> LDS copy (ASYNCcnt)
static __device__ __forceinline__ void async_b128(const bf16* g, const bf16* l) {
  unsigned lo = (unsigned)(uintptr_t)l;                       // LDS byte offset
  unsigned long long ga = (unsigned long long)(uintptr_t)g;
  asm volatile("global_load_async_to_lds_b128 %0, %1, off"
               :: "v"(lo), "v"(ga) : "memory");
}
static __device__ __forceinline__ void wait_async2() {
  asm volatile("s_wait_asynccnt 0x2" ::: "memory");
}
static __device__ __forceinline__ void wait_async0() {
  asm volatile("s_wait_asynccnt 0x0" ::: "memory");
}

// TDM: load a 128x32 bf16 tile (row stride 512 elems) global -> LDS,
// padding LDS by 4 DWORDs (16B) after every 16 DWORDs (one 64B row)
// so rows land at stride 80B == LDA(40) bf16.
static __device__ __forceinline__ void tdm_load_a(unsigned lds_byte, const bf16* gsrc) {
  unsigned long long ga = (unsigned long long)(uintptr_t)gsrc;
  v4u g0; v8u g1;
  g0[0] = 1u;                                   // count=1, user descriptor
  g0[1] = lds_byte;                             // lds_addr
  g0[2] = (unsigned)ga;                         // global_addr lo
  g0[3] = (unsigned)(ga >> 32) | (2u << 30);    // global_addr hi | type=2
  g1[0] = (1u << 16)                            // data_size = 2 bytes
        | (1u << 20)                            // pad_enable
        | (3u << 22)                            // pad_interval: 16 DWORDs
        | (3u << 25);                           // pad_amount:   4 DWORDs
  g1[1] = (512u & 0xffffu) << 16;               // tensor_dim0[15:0] = 512
  g1[2] = (512u & 0xffffu) << 16;               // dim0 hi=0 | tensor_dim1 lo = 512
  g1[3] = 32u << 16;                            // dim1 hi=0 | tile_dim0 = 32
  g1[4] = 128u;                                 // tile_dim1 = 128, tile_dim2 = 0
  g1[5] = 512u;                                 // tensor_dim0_stride lo
  g1[6] = 0u;
  g1[7] = 0u;
  asm volatile("tensor_load_to_lds %0, %1" :: "s"(g0), "s"(g1) : "memory");
}

// -------------------------------------------------- WMMA fragment loads
// A (16x32 bf16), row-major, leading dim ld
__device__ __forceinline__ v16bf frag_a(const bf16* p, int ld, int m0, int kofs, int lane) {
  int row = m0 + (lane & 15);
  int k0  = kofs + ((lane >> 4) << 4);
  const v8bf lo = *(const v8bf*)(p + row * ld + k0);
  const v8bf hi = *(const v8bf*)(p + row * ld + k0 + 8);
  v16bf r;
#pragma unroll
  for (int i = 0; i < 8; ++i) { r[i] = lo[i]; r[i + 8] = hi[i]; }
  return r;
}
// B (32x16 bf16) from N-major source (row = output col, 16 contiguous K)
__device__ __forceinline__ v16bf frag_b(const bf16* p, int ld, int n0, int kofs, int lane) {
  int col = n0 + (lane & 15);
  int kq  = kofs + ((lane >> 4) << 4);
  return *(const v16bf*)(p + col * ld + kq);
}
__device__ __forceinline__ v8f wmma_bf16(v16bf a, v16bf b, v8f c) {
  return __builtin_amdgcn_wmma_f32_16x16x32_bf16(false, a, false, b, (short)0, c, false, false);
}

// ------------------------------------------------------- prepack kernels
__global__ __launch_bounds__(256) void cvt_w(const float* __restrict__ W,
                                             bf16* __restrict__ Wb) {
  size_t base = ((size_t)blockIdx.x * 256 + threadIdx.x) * 8;
  V4<float> a = *(const V4<float>*)(W + base);
  V4<float> c = *(const V4<float>*)(W + base + 4);
  v8bf pk;
#pragma unroll
  for (int j = 0; j < 4; ++j) { pk[j] = f2bf(a.v[j]); pk[4 + j] = f2bf(c.v[j]); }
  *(v8bf*)(Wb + base) = pk;
}

// (B, 512, 2304) f32  ->  (B, 2304, 512) bf16, LDS-tiled 64x64 transpose
__global__ __launch_bounds__(256) void transpose_to_bf16(const float* __restrict__ X,
                                                         bf16* __restrict__ Xt) {
  alignas(16) __shared__ float tl[64 * 68];
  const int t = threadIdx.x;
  const int hwBase = blockIdx.x * 64, cBase = blockIdx.y * 64, b = blockIdx.z;
  const float* Xb = X + ((size_t)b * CD + cBase) * HW + hwBase;
#pragma unroll
  for (int i = 0; i < 4; ++i) {
    int id = t + i * 256;            // 1024 float4 chunks
    int row = id >> 4, col4 = (id & 15) << 2;
    V4<float> v = *(const V4<float>*)(Xb + (size_t)row * HW + col4);
#pragma unroll
    for (int j = 0; j < 4; ++j) tl[row * 68 + col4 + j] = v.v[j];
  }
  __syncthreads();
  bf16* Ob = Xt + ((size_t)b * HW + hwBase) * CD + cBase;
#pragma unroll
  for (int i = 0; i < 2; ++i) {
    int id = t + i * 256;            // 512 8-elem chunks
    int hwr = id >> 3, c8 = (id & 7) << 3;
    v8bf pk;
#pragma unroll
    for (int j = 0; j < 8; ++j) pk[j] = f2bf(tl[(c8 + j) * 68 + hwr]);
    *(v8bf*)(Ob + (size_t)hwr * CD + c8) = pk;
  }
}

// --------------------------------------------------------- GEMM (A Bt^T)
// C(512 x 2304) = A(512x512 bf16 rm) * Bt(2304x512 bf16 N-major)^T + bias
// OUTMODE: 0 = bf16 (B,C,HW) row-major; 1 = bf16 (B,HW,C) packed; 2 = f32 (B,C,HW)
template <int OUTMODE>
__global__ __launch_bounds__(256) void gemm_nt(const bf16* __restrict__ A,
                                               const float* __restrict__ bias,
                                               const bf16* __restrict__ Bt,
                                               void* __restrict__ Yv) {
  constexpr int LDA = 40;            // 32 K + 16B pad (matches TDM padding)
  constexpr int LDB = 48;            // 32 K + pad, keeps 32B alignment
  alignas(32) __shared__ bf16 la[2][128 * LDA];
  alignas(32) __shared__ bf16 lb[2][128 * LDB];

  const int t = threadIdx.x, lane = t & 31, wv = t >> 5;
  const int nBase = blockIdx.x * 128, mBase = blockIdx.y * 128, b = blockIdx.z;
  const bf16* Bb = Bt + (size_t)b * HW * CD;
  const int wm = (wv & 3) * 32, wn = (wv >> 2) * 64;

  const v8f vzero = {0.f, 0.f, 0.f, 0.f, 0.f, 0.f, 0.f, 0.f};
  v8f acc[2][4];
#pragma unroll
  for (int mi = 0; mi < 2; ++mi)
#pragma unroll
    for (int ni = 0; ni < 4; ++ni) acc[mi][ni] = vzero;

  // stage issue: B tile per-lane async b128 (2 instr/wave), A tile via TDM (wave 0)
  auto issueB = [&](int buf, int kk) {
#pragma unroll
    for (int i = 0; i < 2; ++i) {
      int id = t + i * 256;          // 512 chunks of 16B
      int n = id >> 2, c = (id & 3) * 8;
      async_b128(Bb + (size_t)(nBase + n) * CD + kk + c, &lb[buf][n * LDB + c]);
    }
  };

  issueB(0, 0);
  if (wv == 0) tdm_load_a((unsigned)(uintptr_t)&la[0][0], A + (size_t)mBase * CD);

  for (int kk = 0, s = 0; kk < CD; kk += 32, ++s) {
    const int cur = s & 1;
    const bool more = (kk + 32) < CD;
    if (more) {
      issueB(cur ^ 1, kk + 32);
      if (wv == 0)
        tdm_load_a((unsigned)(uintptr_t)&la[cur ^ 1][0],
                   A + (size_t)mBase * CD + kk + 32);
    }
    if (more) wait_async2(); else wait_async0();
    if (wv == 0) {
      if (more) __builtin_amdgcn_s_wait_tensorcnt((short)1);
      else      __builtin_amdgcn_s_wait_tensorcnt((short)0);
    }
    __syncthreads();

    v16bf af0 = frag_a(&la[cur][0], LDA, wm, 0, lane);
    v16bf af1 = frag_a(&la[cur][0], LDA, wm + 16, 0, lane);
    v16bf bg[4];
#pragma unroll
    for (int ni = 0; ni < 4; ++ni) bg[ni] = frag_b(&lb[cur][0], LDB, wn + ni * 16, 0, lane);
#pragma unroll
    for (int ni = 0; ni < 4; ++ni) acc[0][ni] = wmma_bf16(af0, bg[ni], acc[0][ni]);
#pragma unroll
    for (int ni = 0; ni < 4; ++ni) acc[1][ni] = wmma_bf16(af1, bg[ni], acc[1][ni]);
    __syncthreads();
  }

  // epilogue (C layout: VGPR i -> M = m0+i, N = lane&15)
#pragma unroll
  for (int mi = 0; mi < 2; ++mi) {
#pragma unroll
    for (int ni = 0; ni < 4; ++ni) {
      int m0   = mBase + wm + mi * 16 + ((lane >> 4) << 3);
      int ncol = nBase + wn + ni * 16 + (lane & 15);
      if constexpr (OUTMODE == 1) {
        v8bf pk;
#pragma unroll
        for (int i = 0; i < 8; ++i) pk[i] = f2bf(acc[mi][ni][i] + bias[m0 + i]);
        *(v8bf*)((bf16*)Yv + ((size_t)b * HW + ncol) * CD + m0) = pk;
      } else if constexpr (OUTMODE == 0) {
#pragma unroll
        for (int i = 0; i < 8; ++i)
          ((bf16*)Yv)[((size_t)b * CD + m0 + i) * HW + ncol] =
              f2bf(acc[mi][ni][i] + bias[m0 + i]);
      } else {
#pragma unroll
        for (int i = 0; i < 8; ++i)
          ((float*)Yv)[((size_t)b * CD + m0 + i) * HW + ncol] =
              acc[mi][ni][i] + bias[m0 + i];
      }
    }
  }
}

// -------------------------------------------------------- attention core
// block = (batch, head). Q,K: bf16 (B,512,2304); V,O transposed: bf16 (B,2304,512)
__global__ __launch_bounds__(256) void attn64(const bf16* __restrict__ Q,
                                              const bf16* __restrict__ K,
                                              const bf16* __restrict__ Vt,
                                              bf16* __restrict__ Ot) {
  constexpr int LSC = 68, LAT = 72;
  alignas(32) __shared__ float sc[64 * LSC];
  alignas(32) __shared__ bf16  at[64 * LAT];

  const int t = threadIdx.x, lane = t & 31, wv = t >> 5;
  const int bx = blockIdx.x, bb = bx >> 3, hh = bx & 7;
  const size_t qkbase = (size_t)bx * 64 * HW;
  const bf16* Qh = Q + qkbase;
  const bf16* Kh = K + qkbase;
  const bf16* Vh = Vt + (size_t)bb * HW * CD + hh * 64;
  bf16*       Oh = Ot + (size_t)bb * HW * CD + hh * 64;

  const v8f vzero = {0.f, 0.f, 0.f, 0.f, 0.f, 0.f, 0.f, 0.f};

  // phase 1: S = Q K^T over d=2304 (direct from L2; fragment-native layout)
  {
    const int mi  = (wv >> 1) * 16;
    const int nj0 = (wv & 1) * 32;
    v8f s[2] = {vzero, vzero};
    for (int d = 0; d < HW; d += 32) {
      v16bf aq = frag_a(Qh, HW, mi, d, lane);
#pragma unroll
      for (int jj = 0; jj < 2; ++jj) {
        v16bf bk = frag_b(Kh, HW, nj0 + jj * 16, d, lane);
        s[jj] = wmma_bf16(aq, bk, s[jj]);
      }
    }
#pragma unroll
    for (int jj = 0; jj < 2; ++jj) {
      int col = nj0 + jj * 16 + (lane & 15);
      int m0  = mi + ((lane >> 4) << 3);
#pragma unroll
      for (int i = 0; i < 8; ++i) sc[(m0 + i) * LSC + col] = s[jj][i] * ATTN_SCALE;
    }
  }
  __syncthreads();

  if (t < 64) {
    float mx = -3.4e38f;
    for (int j = 0; j < 64; ++j) mx = fmaxf(mx, sc[t * LSC + j]);
    float sum = 0.f;
    for (int j = 0; j < 64; ++j) {
      float e = __expf(sc[t * LSC + j] - mx);
      sc[t * LSC + j] = e;
      sum += e;
    }
    float inv = 1.f / sum;
    for (int j = 0; j < 64; ++j) at[t * LAT + j] = f2bf(sc[t * LSC + j] * inv);
  }
  __syncthreads();

  // phase 2: O^T = (attn * V)^T, V fragments straight from global (N-major)
  v16bf af[4][2];
#pragma unroll
  for (int mi = 0; mi < 4; ++mi)
#pragma unroll
    for (int ks = 0; ks < 2; ++ks) af[mi][ks] = frag_a(at, LAT, mi * 16, ks * 32, lane);

  for (int r = 0; r < 18; ++r) {
    const int n0 = r * 128 + wv * 16;
    v8f oc[4] = {vzero, vzero, vzero, vzero};
#pragma unroll
    for (int ks = 0; ks < 2; ++ks) {
      v16bf bv = frag_b(Vh, CD, n0, ks * 32, lane);
#pragma unroll
      for (int mi = 0; mi < 4; ++mi) oc[mi] = wmma_bf16(af[mi][ks], bv, oc[mi]);
    }
    const int n = n0 + (lane & 15);
#pragma unroll
    for (int mi = 0; mi < 4; ++mi) {
      int m0 = mi * 16 + ((lane >> 4) << 3);
      v8bf pk;
#pragma unroll
      for (int i = 0; i < 8; ++i) pk[i] = f2bf(oc[mi][i]);
      *(v8bf*)(Oh + (size_t)n * CD + m0) = pk;
    }
  }
}

// ------------------------------------------------------------- launcher
extern "C" void kernel_launch(void* const* d_in, const int* in_sizes, int n_in,
                              void* d_out, int out_size, void* d_ws, size_t ws_size,
                              hipStream_t stream) {
  (void)in_sizes; (void)n_in; (void)out_size; (void)ws_size;
  const float* xq = (const float*)d_in[0];
  const float* xk = (const float*)d_in[1];
  const float* xv = (const float*)d_in[2];
  const float* Wq = (const float*)d_in[3];
  const float* bq = (const float*)d_in[4];
  const float* Wk = (const float*)d_in[5];
  const float* bk = (const float*)d_in[6];
  const float* Wv = (const float*)d_in[7];
  const float* bv = (const float*)d_in[8];
  const float* Wo = (const float*)d_in[9];
  const float* bo = (const float*)d_in[10];
  float* out = (float*)d_out;

  const size_t NE = (size_t)NB * CD * HW;      // 18,874,368
  const size_t WE = (size_t)CD * CD;           // 262,144
  bf16* Qb  = (bf16*)d_ws;
  bf16* Kb  = Qb + NE;
  bf16* Vt  = Qb + 2 * NE;
  bf16* Xt  = Qb + 3 * NE;                     // reused as O^T after V gemm
  bf16* Wqb = Qb + 4 * NE;
  bf16* Wkb = Wqb + WE;
  bf16* Wvb = Wkb + WE;
  bf16* Wob = Wvb + WE;

  dim3 blk(256);
  dim3 gW(WE / (256 * 8));                     // 128 blocks
  cvt_w<<<gW, blk, 0, stream>>>(Wq, Wqb);
  cvt_w<<<gW, blk, 0, stream>>>(Wk, Wkb);
  cvt_w<<<gW, blk, 0, stream>>>(Wv, Wvb);
  cvt_w<<<gW, blk, 0, stream>>>(Wo, Wob);

  dim3 gT(HW / 64, CD / 64, NB);               // (36, 8, 16)
  dim3 gG(HW / 128, CD / 128, NB);             // (18, 4, 16)

  transpose_to_bf16<<<gT, blk, 0, stream>>>(xq, Xt);
  gemm_nt<0><<<gG, blk, 0, stream>>>(Wqb, bq, Xt, Qb);
  transpose_to_bf16<<<gT, blk, 0, stream>>>(xk, Xt);
  gemm_nt<0><<<gG, blk, 0, stream>>>(Wkb, bk, Xt, Kb);
  transpose_to_bf16<<<gT, blk, 0, stream>>>(xv, Xt);
  gemm_nt<1><<<gG, blk, 0, stream>>>(Wvb, bv, Xt, Vt);

  attn64<<<dim3(NB * NHEAD), blk, 0, stream>>>(Qb, Kb, Vt, Xt);  // Xt = O^T

  gemm_nt<2><<<gG, blk, 0, stream>>>(Wob, bo, Xt, out);
}
